// Panoptic_spherical_contrastive_loss_19980187861369
// MI455X (gfx1250) — matrix operand order repaired and checked by reading, not compiled, not run on hardware
//
#include <hip/hip_runtime.h>
#include <math.h>

typedef __attribute__((ext_vector_type(16))) _Float16 v16h;
typedef __attribute__((ext_vector_type(8)))  _Float16 v8h;
typedef __attribute__((ext_vector_type(8)))  float    v8f;

namespace {
constexpr int Bn   = 4;
constexpr int Dch  = 32;
constexpr int HWn  = 512 * 512;
constexpr int Cn   = 8;
constexpr int Sg   = 4;              // segments per category
constexpr int Pn   = 8192;           // pixels per segment
constexpr int PITCH = 40;            // halves per pixel row in LDS (16B-aligned rows)
constexpr int NCHUNK = 16;           // chunks per (b,c)
constexpr int CHUNK_PX = Pn / NCHUNK;      // 512 positions
constexpr int NQUAD = CHUNK_PX / 64;       // 8 quad-tiles of 64 positions
// workspace layout (floats)
constexpr int TRIU_OFF = 0;          // B*C*10 = 320
constexpr int RS_OFF   = 320;        // 8 radius err sums
constexpr int RC_OFF   = 328;        // 8 radius counts
constexpr int CU_OFF   = 384;        // chunk channel-sums: B*C*NCHUNK*S*32 = 65536
}

__device__ __forceinline__ int ldsIdx(int s, int px, int d) {
  return (s * 64 + px) * PITCH + d;
}

__global__ void psc_init(float* __restrict__ ws) {
  const int t = threadIdx.x;
  for (int i = t; i < 336; i += 64) ws[i] = 0.f;
}

__global__ __launch_bounds__(32)
void psc_main(const float* __restrict__ outputs, const int* __restrict__ masks,
              float* __restrict__ ws) {
  __shared__ __align__(16) _Float16 lds[4 * 64 * PITCH];
  __shared__ float radS[8];
  __shared__ float radC[8];

  const int lane = threadIdx.x;
  const int blk  = blockIdx.x;
  const int k = blk % NCHUNK;
  const int c = (blk / NCHUNK) % Cn;
  const int b = blk / (NCHUNK * Cn);

  if (lane < 8) { radS[lane] = 0.f; radC[lane] = 0.f; }
  __syncthreads();

  const int hi = lane >> 4;       // half-wave index (WMMA fragment row group)
  const int m  = lane & 15;       // position within 16-tile / output column

  float pairAcc[10];
  float Pre[4];
#pragma unroll
  for (int p = 0; p < 10; ++p) pairAcc[p] = 0.f;
#pragma unroll
  for (int s = 0; s < 4; ++s) Pre[s] = 0.f;

  const int PI[10] = {0, 0, 0, 0, 1, 1, 1, 2, 2, 3};
  const int PJ[10] = {0, 1, 2, 3, 1, 2, 3, 2, 3, 3};

  for (int q = 0; q < NQUAD; ++q) {
    const int pos0 = k * CHUNK_PX + q * 64;

    // ---- stage 4 segments x 64 positions x 32 channels into LDS as f16
#pragma unroll
    for (int s = 0; s < 4; ++s) {
      const size_t pg0 = (size_t)(c * Sg + s) * Pn + pos0;
#pragma unroll
      for (int d = 0; d < Dch; ++d) {
        const float2 g = *(const float2*)(outputs +
                           ((size_t)(b * Dch + d)) * HWn + pg0 + 2 * lane);
        lds[ldsIdx(s, 2 * lane,     d)] = (_Float16)g.x;
        lds[ldsIdx(s, 2 * lane + 1, d)] = (_Float16)g.y;
      }
    }
    __syncthreads();

    // ---- per-pixel norm, radius partials, in-place normalize (vector DS ops)
#pragma unroll
    for (int s = 0; s < 4; ++s) {
#pragma unroll
      for (int u = 0; u < 2; ++u) {
        const int pp = 2 * lane + u;
        v8h h[4];
#pragma unroll
        for (int r = 0; r < 4; ++r)
          h[r] = *(const v8h*)&lds[ldsIdx(s, pp, r * 8)];
        float a0 = 0.f, a1 = 0.f, a2 = 0.f, a3 = 0.f;
#pragma unroll
        for (int t = 0; t < 8; ++t) {
          const float x0 = (float)h[0][t], x1 = (float)h[1][t];
          const float x2 = (float)h[2][t], x3 = (float)h[3][t];
          a0 = fmaf(x0, x0, a0); a1 = fmaf(x1, x1, a1);
          a2 = fmaf(x2, x2, a2); a3 = fmaf(x3, x3, a3);
        }
        const float nrm = sqrtf((a0 + a1) + (a2 + a3));
        const size_t pg = (size_t)(c * Sg + s) * Pn + pos0 + pp;
        const int cat = masks[(size_t)(b * 3 + 1) * HWn + pg] & 7;
        const float dr = nrm - (1.0f + (float)cat);
        atomicAdd(&radS[cat], dr * dr);
        atomicAdd(&radC[cat], 1.0f);
        const _Float16 hinv = (_Float16)(1.0f / (nrm + 1e-6f));
#pragma unroll
        for (int r = 0; r < 4; ++r) {
          v8h hh = h[r];
#pragma unroll
          for (int t = 0; t < 8; ++t) hh[t] *= hinv;
          *(v8h*)&lds[ldsIdx(s, pp, r * 8)] = hh;
        }
      }
    }
    __syncthreads();

    // ---- per 16-position tile: rank-1 cross-tile scan + WMMA triangular grams
#pragma unroll
    for (int tt = 0; tt < 4; ++tt) {
      const int tbase = tt * 16;

      // tile channel-sums, distributed lane <-> channel (d = lane)
      // batched independent loads, then tree reduction (no per-load waits)
      float tu[4];
#pragma unroll
      for (int s = 0; s < 4; ++s) {
        float tmp[16];
#pragma unroll
        for (int mm = 0; mm < 16; ++mm)
          tmp[mm] = (float)lds[ldsIdx(s, tbase + mm, lane)];
        float t0 = 0.f, t1 = 0.f, t2 = 0.f, t3 = 0.f;
#pragma unroll
        for (int mm = 0; mm < 16; mm += 4) {
          t0 += tmp[mm];     t1 += tmp[mm + 1];
          t2 += tmp[mm + 2]; t3 += tmp[mm + 3];
        }
        tu[s] = (t0 + t1) + (t2 + t3);
      }
      // earlier-tiles x this-tile rank-1 contributions (per-lane partial dots)
#pragma unroll
      for (int p = 0; p < 10; ++p)
        pairAcc[p] += Pre[PI[p]] * tu[PJ[p]];
#pragma unroll
      for (int s = 0; s < 4; ++s) Pre[s] += tu[s];

      // load all fragments once per tile, then 10 back-to-back WMMAs
      const int px = tbase + m;
      v16h Af[4], Bf[4];
#pragma unroll
      for (int s = 0; s < 4; ++s) {
        // A fragment (16-bit A 16x32): lane row M=m; K = {hi*8.., 16+hi*8..}
        const v8h a0 = *(const v8h*)&lds[ldsIdx(s, px, hi * 8)];
        const v8h a1 = *(const v8h*)&lds[ldsIdx(s, px, 16 + hi * 8)];
        // B fragment (16-bit B 32x16): lane col N=m; K = {hi*8.., 16+hi*8..}
        const v8h b0 = *(const v8h*)&lds[ldsIdx(s, px, hi * 8)];
        const v8h b1 = *(const v8h*)&lds[ldsIdx(s, px, 16 + hi * 8)];
#pragma unroll
        for (int t = 0; t < 8; ++t) {
          Af[s][t] = a0[t]; Af[s][t + 8] = a1[t];
          Bf[s][t] = b0[t]; Bf[s][t + 8] = b1[t];
        }
      }
      v8f acc[10];
#pragma unroll
      for (int p = 0; p < 10; ++p) {
        v8f z = {};
        acc[p] = __builtin_amdgcn_wmma_f32_16x16x32_f16(
                     false, Af[PI[p]], false, Bf[PJ[p]], (short)0, z,
                     false, false);
      }
      // keep triu (p<=q): row M = r + 8*hi, col N = m
#pragma unroll
      for (int p = 0; p < 10; ++p)
#pragma unroll
        for (int r = 0; r < 8; ++r)
          pairAcc[p] += ((r + 8 * hi) <= m) ? acc[p][r] : 0.f;
    }
    __syncthreads();
  }

  // ---- chunk epilogue
  const int bc = b * Cn + c;
#pragma unroll
  for (int s = 0; s < 4; ++s)
    ws[CU_OFF + (((size_t)bc * NCHUNK + k) * 4 + s) * 32 + lane] = Pre[s];

#pragma unroll
  for (int p = 0; p < 10; ++p) {
    float v = pairAcc[p];
#pragma unroll
    for (int o = 16; o > 0; o >>= 1) v += __shfl_xor(v, o, 32);
    if (lane == 0) atomicAdd(&ws[TRIU_OFF + bc * 10 + p], v);
  }
  __syncthreads();
  if (lane < 8) {
    atomicAdd(&ws[RS_OFF + lane], radS[lane]);
    atomicAdd(&ws[RC_OFF + lane], radC[lane]);
  }
}

__global__ __launch_bounds__(32)
void psc_final(float* __restrict__ ws, float* __restrict__ out) {
  const int lane = threadIdx.x;
  const int PI[10] = {0, 0, 0, 0, 1, 1, 1, 2, 2, 3};
  const int PJ[10] = {0, 1, 2, 3, 1, 2, 3, 2, 3, 3};

  // cross-chunk rank-1 corrections: sum_{k1<k2} chunkU_i[k1] . chunkU_j[k2]
  for (int bc = 0; bc < 32; ++bc) {
    float Pre[4] = {0.f, 0.f, 0.f, 0.f};
    float pl[10];
#pragma unroll
    for (int p = 0; p < 10; ++p) pl[p] = 0.f;
    for (int kk = 0; kk < NCHUNK; ++kk) {
      float cu[4];
#pragma unroll
      for (int s = 0; s < 4; ++s)
        cu[s] = ws[CU_OFF + (((size_t)bc * NCHUNK + kk) * 4 + s) * 32 + lane];
#pragma unroll
      for (int p = 0; p < 10; ++p) pl[p] += Pre[PI[p]] * cu[PJ[p]];
#pragma unroll
      for (int s = 0; s < 4; ++s) Pre[s] += cu[s];
    }
#pragma unroll
    for (int p = 0; p < 10; ++p) {
      float v = pl[p];
#pragma unroll
      for (int o = 16; o > 0; o >>= 1) v += __shfl_xor(v, o, 32);
      if (lane == 0) ws[TRIU_OFF + bc * 10 + p] += v;
    }
  }
  __syncthreads();

  if (lane == 0) {
    const float count = 8192.0f * 8193.0f * 0.5f;   // P*(P+1)/2
    float tot = 0.f;
    for (int bc = 0; bc < 32; ++bc) {
#pragma unroll
      for (int p = 0; p < 10; ++p) {
        const float t = ws[TRIU_OFF + bc * 10 + p];
        tot += (PI[p] == PJ[p]) ? (1.0f - t / count)   // positives
                                : (t / count + 2.0f);  // negatives, -MARGIN=+2
      }
    }
    const float sim = tot / 320.0f;  // B*C*(S + S*(S-1)/2)
    float rl = 0.f;
    for (int cc = 1; cc < 8; ++cc) rl += ws[RS_OFF + cc] / ws[RC_OFF + cc];
    rl *= (1.0f / 7.0f);
    out[0] = 0.5f * rl + 0.5f * sim;
  }
}

extern "C" void kernel_launch(void* const* d_in, const int* in_sizes, int n_in,
                              void* d_out, int out_size, void* d_ws, size_t ws_size,
                              hipStream_t stream) {
  const float* outputs = (const float*)d_in[0];
  const int*   masks   = (const int*)d_in[1];
  float* ws  = (float*)d_ws;
  float* out = (float*)d_out;
  (void)in_sizes; (void)n_in; (void)out_size; (void)ws_size;

  psc_init<<<dim3(1), dim3(64), 0, stream>>>(ws);
  psc_main<<<dim3(Bn * Cn * NCHUNK), dim3(32), 0, stream>>>(outputs, masks, ws);
  psc_final<<<dim3(1), dim3(32), 0, stream>>>(ws, out);
}
